// ResidualAttentionBlock_2637109920222
// MI455X (gfx1250) — matrix operand order repaired and checked by reading, not compiled, not run on hardware
//
#include <hip/hip_runtime.h>
#include <math.h>

typedef __attribute__((ext_vector_type(16))) _Float16 v16h;
typedef __attribute__((ext_vector_type(8)))  _Float16 v8h;
typedef __attribute__((ext_vector_type(8)))  float    v8f;

#define NB   8
#define SEQ  2048
#define DIN  768
#define DH   1024

// ---------------------------------------------------------------------------
// CDNA5 async copy: global -> LDS, 16B per lane, tracked by ASYNCcnt.
// (CDNA5 ISA opcode 98 / CDNA5 ISA)
// ---------------------------------------------------------------------------
__device__ inline void async_copy_b128(void* lds, const void* gaddr) {
    unsigned l = (unsigned)(size_t)lds;  // flat LDS addr low 32 bits == LDS offset
    asm volatile("global_load_async_to_lds_b128 %0, %1, off"
                 :: "v"(l), "v"(gaddr)
                 : "memory");
}
__device__ inline void wait_asynccnt0() {
    asm volatile("s_wait_asynccnt 0x0" ::: "memory");
}

// ---------------------------------------------------------------------------
// WMMA fragment loads from LDS (CDNA5 ISA layouts).
// A (16x32 f16): lane L -> M=L&15, halves 0-7 = K kg*8.., 8-15 = K 16+kg*8..
// B (32x16 f16): lane -> col lane&15, halves 0-7 = K kg*8.., 8-15 = K 16+kg*8..
// ---------------------------------------------------------------------------
__device__ inline v16h frag_a(const _Float16 (*S)[40], int base, int lane) {
    int row = base + (lane & 15);
    int kg  = (lane >> 4) * 8;
    v8h lo = *(const v8h*)&S[row][kg];
    v8h hi = *(const v8h*)&S[row][16 + kg];
    v16h r;
#pragma unroll
    for (int j = 0; j < 8; ++j) { r[j] = lo[j]; r[j + 8] = hi[j]; }
    return r;
}
__device__ inline v16h frag_b(const _Float16 (*S)[40], int base, int lane) {
    int col = base + (lane & 15);
    int kh  = (lane >> 4) * 8;
    v8h lo = *(const v8h*)&S[col][kh];
    v8h hi = *(const v8h*)&S[col][16 + kh];
    v16h r;
#pragma unroll
    for (int j = 0; j < 8; ++j) { r[j] = lo[j]; r[j + 8] = hi[j]; }
    return r;
}

// ---------------------------------------------------------------------------
// C[M,N] = A[M,K] @ B[N,K]^T  (+bias / relu / f32-accumulate / f16 / f16^T out)
// Block tile 128x128, Ktile 32, 256 threads = 8 waves (2x4), wave tile 64x32.
// Double-buffered LDS fed by async b128 copies. Batched via blockIdx.z.
// flags: 1=relu  2=accumulate into outF  4=store f16 transposed (seq-major)
// ---------------------------------------------------------------------------
__launch_bounds__(256)
__global__ void gemm_f16_nt(const _Float16* __restrict__ A,
                            const _Float16* __restrict__ Bm,
                            const float* __restrict__ bias,
                            _Float16* __restrict__ outH,
                            float* __restrict__ outF,
                            int N, int K,
                            long long sA, long long sB, long long sC,
                            float alpha, int flags) {
    __shared__ __align__(16) _Float16 As[2][128][40];
    __shared__ __align__(16) _Float16 Bs[2][128][40];
    const int tid  = threadIdx.x;
    const int lane = tid & 31;
    const int wid  = tid >> 5;
    const int wRow = (wid >> 2) * 64;   // 2 wave-rows
    const int wCol = (wid & 3) * 32;    // 4 wave-cols
    const int rowBase = blockIdx.y * 128;
    const int colBase = blockIdx.x * 128;
    const int z = blockIdx.z;
    const _Float16* Ab = A  + (size_t)z * sA + (size_t)rowBase * K;
    const _Float16* Bb = Bm + (size_t)z * sB + (size_t)colBase * K;

    // per-thread copy chunks: 128 rows x 4 b128-chunks = 512 per matrix
    const int r0  = tid >> 2;          // 0..63
    const int kc0 = (tid & 3) * 8;     // half offset within Ktile

    auto issue_tile = [&](int buf, int k0) {
        async_copy_b128(&As[buf][r0][kc0],      Ab + (size_t)r0 * K + k0 + kc0);
        async_copy_b128(&As[buf][r0 + 64][kc0], Ab + (size_t)(r0 + 64) * K + k0 + kc0);
        async_copy_b128(&Bs[buf][r0][kc0],      Bb + (size_t)r0 * K + k0 + kc0);
        async_copy_b128(&Bs[buf][r0 + 64][kc0], Bb + (size_t)(r0 + 64) * K + k0 + kc0);
    };

    v8f acc[4][2] = {};
    int buf = 0;
    issue_tile(0, 0);
    for (int k0 = 0; k0 < K; k0 += 32) {
        wait_asynccnt0();      // this thread's copies for tile(buf) landed
        __syncthreads();       // everyone's landed; prior readers of buf^1 done
        if (k0 + 32 < K) issue_tile(buf ^ 1, k0 + 32);  // overlap with compute

        v16h af[4], bf[2];
#pragma unroll
        for (int m = 0; m < 4; ++m) af[m] = frag_a(As[buf], wRow + m * 16, lane);
#pragma unroll
        for (int n = 0; n < 2; ++n) bf[n] = frag_b(Bs[buf], wCol + n * 16, lane);
#pragma unroll
        for (int m = 0; m < 4; ++m)
#pragma unroll
            for (int n = 0; n < 2; ++n)
                acc[m][n] = __builtin_amdgcn_wmma_f32_16x16x32_f16(
                    false, af[m], false, bf[n], (short)0, acc[m][n], false, false);
        buf ^= 1;
    }

    // epilogue: f32 C/D layout -> VGPR r: M = r + (lane>=16 ? 8:0), N = lane&15
#pragma unroll
    for (int m = 0; m < 4; ++m) {
#pragma unroll
        for (int n = 0; n < 2; ++n) {
            int row0 = rowBase + wRow + m * 16 + ((lane >> 4) << 3);
            int col  = colBase + wCol + n * 16 + (lane & 15);
            float bv = bias ? bias[col] : 0.0f;
#pragma unroll
            for (int r = 0; r < 8; ++r) {
                int   row = row0 + r;
                float v   = acc[m][n][r] * alpha + bv;
                if (flags & 1) v = fmaxf(v, 0.0f);
                if (flags & 4) {
                    // v-projection: store transposed vt[b][col][s], s=row%SEQ
                    size_t t = ((size_t)(row >> 11) * DH + col) * SEQ + (row & (SEQ - 1));
                    outH[t] = (_Float16)v;
                } else {
                    size_t idx = (size_t)z * sC + (size_t)row * N + col;
                    if (flags & 2) outF[idx] += v;
                    else {
                        if (outF) outF[idx] = v;
                        if (outH) outH[idx] = (_Float16)v;
                    }
                }
            }
        }
    }
}

// ---------------------------------------------------------------------------
// row softmax: 1 block (256 threads) per row of n=2048, f32 in, f16 out
// ---------------------------------------------------------------------------
__launch_bounds__(256)
__global__ void softmax_rows(const float* __restrict__ scores,
                             _Float16* __restrict__ attn, int n) {
    const long long row = blockIdx.x;
    const float* s = scores + row * (long long)n;
    _Float16* o = attn + row * (long long)n;
    __shared__ float red[256];
    const int t = threadIdx.x;
    const int per = n >> 8;

    float m = -INFINITY;
    for (int j = 0; j < per; ++j) m = fmaxf(m, s[t + j * 256]);
    red[t] = m;
    __syncthreads();
    for (int st = 128; st > 0; st >>= 1) {
        if (t < st) red[t] = fmaxf(red[t], red[t + st]);
        __syncthreads();
    }
    const float rm = red[0];
    __syncthreads();

    float ex[16];
    float sum = 0.0f;
    for (int j = 0; j < per; ++j) {
        ex[j] = __expf(s[t + j * 256] - rm);
        sum += ex[j];
    }
    red[t] = sum;
    __syncthreads();
    for (int st = 128; st > 0; st >>= 1) {
        if (t < st) red[t] += red[t + st];
        __syncthreads();
    }
    const float inv = 1.0f / red[0];
    for (int j = 0; j < per; ++j) o[t + j * 256] = (_Float16)(ex[j] * inv);
}

__global__ void cvt_f32_f16(const float* __restrict__ in,
                            _Float16* __restrict__ out, long long n) {
    long long i = (long long)blockIdx.x * blockDim.x + threadIdx.x;
    long long stride = (long long)gridDim.x * blockDim.x;
    for (; i < n; i += stride) out[i] = (_Float16)in[i];
}

// transpose + narrow: out[n][k] = (f16) in[k][n]   (weights, done once)
__launch_bounds__(256)
__global__ void cvt_t_f32_f16(const float* __restrict__ in,
                              _Float16* __restrict__ out, int K, int N) {
    __shared__ float t[32][33];
    const int k0 = blockIdx.y * 32, n0 = blockIdx.x * 32;
    const int tx = threadIdx.x, ty = threadIdx.y;  // 32 x 8
#pragma unroll
    for (int j = 0; j < 32; j += 8)
        t[ty + j][tx] = in[(size_t)(k0 + ty + j) * N + n0 + tx];
    __syncthreads();
#pragma unroll
    for (int j = 0; j < 32; j += 8)
        out[(size_t)(n0 + ty + j) * K + k0 + tx] = (_Float16)t[tx][ty + j];
}

// ---------------------------------------------------------------------------
extern "C" void kernel_launch(void* const* d_in, const int* in_sizes, int n_in,
                              void* d_out, int out_size, void* d_ws, size_t ws_size,
                              hipStream_t stream) {
    (void)in_sizes; (void)n_in; (void)out_size; (void)ws_size;
    const float* x  = (const float*)d_in[0];
    const float* W1 = (const float*)d_in[1];
    const float* b1 = (const float*)d_in[2];
    const float* Wq = (const float*)d_in[3];
    const float* bq = (const float*)d_in[4];
    const float* Wk = (const float*)d_in[5];
    const float* bk = (const float*)d_in[6];
    const float* Wv = (const float*)d_in[7];
    const float* bv = (const float*)d_in[8];
    const float* W2 = (const float*)d_in[9];
    const float* b2 = (const float*)d_in[10];
    const float* Ws = (const float*)d_in[11];
    const float* bs = (const float*)d_in[12];

    const long long MS = (long long)NB * SEQ;  // 16384 rows flattened

    char* ws = (char*)d_ws;
    size_t off = 0;
    auto alloc = [&](size_t bytes) -> void* {
        void* p = (void*)(ws + off);
        off += (bytes + 255) & ~(size_t)255;
        return p;
    };
    _Float16* xh  = (_Float16*)alloc((size_t)MS * DIN * 2);
    _Float16* w1t = (_Float16*)alloc((size_t)DH * DIN * 2);  // [DH][DIN]
    _Float16* wst = (_Float16*)alloc((size_t)DH * DIN * 2);
    _Float16* wqt = (_Float16*)alloc((size_t)DH * DH * 2);
    _Float16* wkt = (_Float16*)alloc((size_t)DH * DH * 2);
    _Float16* wvt = (_Float16*)alloc((size_t)DH * DH * 2);
    _Float16* w2t = (_Float16*)alloc((size_t)DH * DH * 2);
    _Float16* hh  = (_Float16*)alloc((size_t)MS * DH * 2);
    _Float16* qh  = (_Float16*)alloc((size_t)MS * DH * 2);
    _Float16* kh  = (_Float16*)alloc((size_t)MS * DH * 2);
    _Float16* vt  = (_Float16*)alloc((size_t)MS * DH * 2);  // [NB][DH][SEQ]
    float*    sc  = (float*)   alloc((size_t)NB * SEQ * SEQ * 4);
    _Float16* ath = (_Float16*)alloc((size_t)NB * SEQ * SEQ * 2);
    float*    resf= (float*)   alloc((size_t)MS * DH * 4);
    _Float16* h2h = (_Float16*)alloc((size_t)MS * DH * 2);

    // 1) convert x; transpose+convert all weights (B operands are N x K)
    cvt_f32_f16<<<1024, 256, 0, stream>>>(x, xh, MS * (long long)DIN);
    dim3 tb(32, 8);
    cvt_t_f32_f16<<<dim3(DH / 32, DIN / 32), tb, 0, stream>>>(W1, w1t, DIN, DH);
    cvt_t_f32_f16<<<dim3(DH / 32, DIN / 32), tb, 0, stream>>>(Ws, wst, DIN, DH);
    cvt_t_f32_f16<<<dim3(DH / 32, DH / 32),  tb, 0, stream>>>(Wq, wqt, DH, DH);
    cvt_t_f32_f16<<<dim3(DH / 32, DH / 32),  tb, 0, stream>>>(Wk, wkt, DH, DH);
    cvt_t_f32_f16<<<dim3(DH / 32, DH / 32),  tb, 0, stream>>>(Wv, wvt, DH, DH);
    cvt_t_f32_f16<<<dim3(DH / 32, DH / 32),  tb, 0, stream>>>(W2, w2t, DH, DH);

    const dim3 blk(256);
    // 2) h = relu(x @ W1 + b1)
    gemm_f16_nt<<<dim3(DH / 128, MS / 128, 1), blk, 0, stream>>>(
        xh, w1t, b1, hh, nullptr, DH, DIN, 0, 0, 0, 1.0f, 1);
    // 3) q, k; v written transposed (vt[b][d][s]) for the attn@v GEMM
    gemm_f16_nt<<<dim3(DH / 128, MS / 128, 1), blk, 0, stream>>>(
        hh, wqt, bq, qh, nullptr, DH, DH, 0, 0, 0, 1.0f, 0);
    gemm_f16_nt<<<dim3(DH / 128, MS / 128, 1), blk, 0, stream>>>(
        hh, wkt, bk, kh, nullptr, DH, DH, 0, 0, 0, 1.0f, 0);
    gemm_f16_nt<<<dim3(DH / 128, MS / 128, 1), blk, 0, stream>>>(
        hh, wvt, bv, vt, nullptr, DH, DH, 0, 0, 0, 1.0f, 4);
    // 4) shortcut: resf = x @ Ws + bs (f32)
    gemm_f16_nt<<<dim3(DH / 128, MS / 128, 1), blk, 0, stream>>>(
        xh, wst, bs, nullptr, resf, DH, DIN, 0, 0, 0, 1.0f, 0);
    // 5) scores = (q @ k^T) / sqrt(DH), batched — k is already the NT operand
    gemm_f16_nt<<<dim3(SEQ / 128, SEQ / 128, NB), blk, 0, stream>>>(
        qh, kh, nullptr, nullptr, sc, SEQ, DH,
        (long long)SEQ * DH, (long long)SEQ * DH, (long long)SEQ * SEQ,
        0.03125f /* 1/sqrt(1024) */, 0);
    // 6) attn = softmax(scores) -> f16
    softmax_rows<<<NB * SEQ, 256, 0, stream>>>(sc, ath, SEQ);
    // 7) resf += attn @ v   (B = vt[b] is DH x SEQ row-major = v^T)
    gemm_f16_nt<<<dim3(DH / 128, SEQ / 128, NB), blk, 0, stream>>>(
        ath, vt, nullptr, nullptr, resf, DH, SEQ,
        (long long)SEQ * SEQ, (long long)DH * SEQ, (long long)SEQ * DH,
        1.0f, 2);
    // 8) h2 = f16(resf)
    cvt_f32_f16<<<1024, 256, 0, stream>>>(resf, h2h, MS * (long long)DH);
    // 9) out = relu(h2 @ W2 + b2) -> f32 d_out
    gemm_f16_nt<<<dim3(DH / 128, MS / 128, 1), blk, 0, stream>>>(
        h2h, w2t, b2, nullptr, (float*)d_out, DH, DH, 0, 0, 0, 1.0f, 1);
}
